// HiResPrecipNet_CNN_GNN_19378892439830
// MI455X (gfx1250) — matrix operand order repaired and checked by reading, not compiled, not run on hardware
//
#include <hip/hip_runtime.h>
#include <hip/hip_bf16.h>

// ---------------------------------------------------------------------------
// HiResPrecipNet (CNN + GATv2 GNN) for MI455X / gfx1250.
// Dense projections (GATv2 Wl/Wr, predictor MLP) run on v_wmma_f32_16x16x32_f16
// (f32 activations converted to f16 in LDS, f32 accumulate). LDS tiles are
// K-contiguous for BOTH A and B fragments (B stored transposed) with 144-byte
// padded row stride, so fragment loads lower to ds_load_b128 without bank
// conflicts. Edge softmax/aggregation uses gathers + L2 float atomics;
// segment-max uses the monotone ordered-uint atomicMax trick. BatchNorm batch
// stats: one block per channel.
// ---------------------------------------------------------------------------

#define N_LOW   4000
#define N_UP    100000
#define N_HIGH  100000
#define E_UP    500000
#define E_UH    400000
#define E_HIGH  500000
#define HIDV    32
#define EPSBN   1e-5f
#define SLOPE   0.2f
#define CDIV(a,b) (((a)+(b)-1)/(b))

typedef __attribute__((ext_vector_type(16))) _Float16 v16h;
typedef __attribute__((ext_vector_type(8)))  float    v8f;

// ----------------------------- WMMA GEMM -----------------------------------
// One workgroup per 16-row strip of Y = X[M,K] @ W[K,N] + bias[N].
// blockDim.x = 32 * (N/16); wave w computes columns [16w, 16w+16).
// K <= 64, N in {16,32,64}. A and B staged in LDS as f16 (B transposed so the
// per-lane fragment is K-contiguous), f32 accumulate.
#define KMAX 64
#define TSTR 72   // padded LDS row stride in halves (144B: 16B-aligned, breaks
                  // the 256B bank-period conflict pattern of a 128B stride)
__global__ __launch_bounds__(128)
void gemm16_wmma(const float* __restrict__ X, const float* __restrict__ W,
                 const float* __restrict__ bias, float* __restrict__ Y,
                 int M, int K, int N, int relu)
{
    __shared__ __align__(16) _Float16 As[16 * TSTR];
    __shared__ __align__(16) _Float16 Bs[64 * TSTR];   // [n][k] (transposed)
    const int tid  = threadIdx.x;
    const int nthr = blockDim.x;
    const int wave = tid >> 5;
    const int lane = tid & 31;
    const int m0 = blockIdx.x * 16;
    const int Kp = (K + 31) & ~31;           // pad K to multiple of 32 (<=64)

    for (int idx = tid; idx < 16 * Kp; idx += nthr) {
        int r = idx / Kp, c = idx % Kp;
        float v = 0.f;
        int gr = m0 + r;
        if (gr < M && c < K) v = X[(size_t)gr * K + c];
        As[r * TSTR + c] = (_Float16)v;
    }
    for (int idx = tid; idx < N * Kp; idx += nthr) {
        int c = idx / Kp, r = idx % Kp;       // c = output column, r = K index
        float v = (r < K) ? W[(size_t)r * N + c] : 0.f;
        Bs[c * TSTR + r] = (_Float16)v;
    }
    __syncthreads();

    const int n0 = wave * 16;
    v8f acc = {};
    const int rowA  = lane & 15;
    const int colB  = lane & 15;
    const int ahalf = (lane < 16) ? 0 : 16;   // A frag: contiguous K striping
    const int bhalf = (lane < 16) ? 0 : 16;   // B frag: contiguous K striping
    for (int kk = 0; kk < Kp; kk += 32) {
        v16h a, b;
#pragma unroll
        for (int i = 0; i < 16; ++i)
            a[i] = As[rowA * TSTR + kk + ahalf + i];
#pragma unroll
        for (int i = 0; i < 16; ++i)
            b[i] = Bs[(n0 + colB) * TSTR + kk + bhalf + i];
        acc = __builtin_amdgcn_wmma_f32_16x16x32_f16(false, a, false, b,
                                                     (short)0, acc, false, false);
    }

    const int col   = n0 + colB;
    const int rbase = m0 + ((lane < 16) ? 0 : 8);
    const float bv  = bias ? bias[col] : 0.f;
#pragma unroll
    for (int r = 0; r < 8; ++r) {
        int row = rbase + r;
        if (row < M) {
            float v = acc[r] + bv;
            if (relu) v = fmaxf(v, 0.f);
            Y[(size_t)row * N + col] = v;
        }
    }
}

// ----------------------------- CNN kernels ----------------------------------
__global__ void dwconv2d_k(const float* __restrict__ x, const float* __restrict__ w,
                           const float* __restrict__ b, float* __restrict__ y, int N)
{
    int idx = blockIdx.x * blockDim.x + threadIdx.x;
    if (idx >= N * 5 * 25) return;
    int j = idx % 5, i = (idx / 5) % 5, c = (idx / 25) % 5, n = idx / 125;
    float acc = b[c];
    for (int di = 0; di < 3; ++di) {
        int ii = i + di - 1; if (ii < 0 || ii >= 5) continue;
        for (int dj = 0; dj < 3; ++dj) {
            int jj = j + dj - 1; if (jj < 0 || jj >= 5) continue;
            acc += x[((n * 5 + c) * 5 + ii) * 5 + jj] * w[(c * 3 + di) * 3 + dj];
        }
    }
    y[idx] = acc;
}

__global__ void maxpool2d_k(const float* __restrict__ x, float* __restrict__ y, int N)
{
    int idx = blockIdx.x * blockDim.x + threadIdx.x;
    if (idx >= N * 45) return;
    int oj = idx % 3, oi = (idx / 3) % 3, c = (idx / 9) % 5, n = idx / 45;
    float m = -3.4e38f;
    for (int ki = 0; ki < 2; ++ki) {
        int ii = 2 * oi - 1 + ki; if (ii < 0 || ii >= 5) continue;
        for (int kj = 0; kj < 2; ++kj) {
            int jj = 2 * oj - 1 + kj; if (jj < 0 || jj >= 5) continue;
            m = fmaxf(m, x[((n * 5 + c) * 5 + ii) * 5 + jj]);
        }
    }
    y[n * 45 + c * 9 + oi * 3 + oj] = m;
}

__global__ void conv1d_k(const float* __restrict__ x, const float* __restrict__ w,
                         const float* __restrict__ b, float* __restrict__ y,
                         int N, int Ci, int Co, int L)
{
    int idx = blockIdx.x * blockDim.x + threadIdx.x;
    if (idx >= N * Co * L) return;
    int t = idx % L, co = (idx / L) % Co, n = idx / (L * Co);
    float acc = b[co];
    for (int ci = 0; ci < Ci; ++ci) {
        const float* xp = x + ((size_t)n * Ci + ci) * L;
        const float* wp = w + (co * Ci + ci) * 3;
        if (t > 0)     acc += xp[t - 1] * wp[0];
        acc += xp[t] * wp[1];
        if (t < L - 1) acc += xp[t + 1] * wp[2];
    }
    y[idx] = acc;
}

__global__ void maxpool1d_k(const float* __restrict__ x, float* __restrict__ y, int N)
{
    int idx = blockIdx.x * blockDim.x + threadIdx.x;   // over N*25*23
    if (idx >= N * 25 * 23) return;
    int ot = idx % 23, c = (idx / 23) % 25, n = idx / (23 * 25);
    float m = -3.4e38f;
    for (int k = 0; k < 2; ++k) {
        int tt = 2 * ot - 1 + k;
        if (tt >= 0 && tt < 45) m = fmaxf(m, x[((size_t)n * 25 + c) * 45 + tt]);
    }
    y[idx] = m;   // flat [N*25, 23] row-major == idx order
}

// ----------------------------- BatchNorm ------------------------------------
// Layout [N, C, S]; per-channel stats over N*S elements. One block per channel.
__global__ __launch_bounds__(256)
void bn_stats_strided(const float* __restrict__ x, int N, int C, int S,
                      float* __restrict__ meanv, float* __restrict__ rstdv)
{
    int c = blockIdx.x;
    int total = N * S;
    float s = 0.f, s2 = 0.f;
    for (int i = threadIdx.x; i < total; i += 256) {
        int n = i / S, sp = i % S;
        float v = x[((size_t)n * C + c) * S + sp];
        s += v; s2 += v * v;
    }
    __shared__ float sh[512];
    sh[threadIdx.x] = s; sh[256 + threadIdx.x] = s2;
    __syncthreads();
    for (int st = 128; st > 0; st >>= 1) {
        if ((int)threadIdx.x < st) {
            sh[threadIdx.x]       += sh[threadIdx.x + st];
            sh[256 + threadIdx.x] += sh[256 + threadIdx.x + st];
        }
        __syncthreads();
    }
    if (threadIdx.x == 0) {
        float mu  = sh[0] / total;
        float var = sh[256] / total - mu * mu;
        meanv[c] = mu;
        rstdv[c] = rsqrtf(var + EPSBN);
    }
}

__global__ void bn_apply_strided(const float* __restrict__ x, float* __restrict__ y,
                                 int N, int C, int S,
                                 const float* __restrict__ meanv, const float* __restrict__ rstdv,
                                 const float* __restrict__ gamma, const float* __restrict__ beta,
                                 int relu)
{
    int idx = blockIdx.x * blockDim.x + threadIdx.x;
    int total = N * C * S;
    if (idx >= total) return;
    int c = (idx / S) % C;
    float v = gamma[c] * (x[idx] - meanv[c]) * rstdv[c] + beta[c];
    if (relu) v = fmaxf(v, 0.f);
    y[idx] = v;
}

// ----------------------------- GATv2 edge kernels ---------------------------
__device__ __forceinline__ unsigned fkey(float f) {
    unsigned u = __float_as_uint(f);
    return (u & 0x80000000u) ? ~u : (u | 0x80000000u);
}
__device__ __forceinline__ float funkey(unsigned k) {
    return (k & 0x80000000u) ? __uint_as_float(k & 0x7FFFFFFFu) : __uint_as_float(~k);
}

__global__ void degree_k(const int* __restrict__ dst, int E, int nloops,
                         float* __restrict__ cnt)
{
    int idx = blockIdx.x * blockDim.x + threadIdx.x;
    if (idx >= E + nloops) return;
    int d = (idx < E) ? dst[idx] : (idx - E);
    atomicAdd(&cnt[d], 1.0f);
}

__global__ void gat_edge_logit(const int* __restrict__ src, const int* __restrict__ dst,
                               int E, int nloops,
                               const float* __restrict__ xl, const float* __restrict__ xr,
                               const float* __restrict__ att, int H, int D,
                               float* __restrict__ logit, unsigned* __restrict__ mx)
{
    int idx = blockIdx.x * blockDim.x + threadIdx.x;
    int total = (E + nloops) * H;
    if (idx >= total) return;
    int e = idx / H, h = idx % H;
    int s = (e < E) ? src[e] : (e - E);
    int d = (e < E) ? dst[e] : (e - E);
    int HD = H * D;
    const float* pl = xl + (size_t)s * HD + h * D;
    const float* pr = xr + (size_t)d * HD + h * D;
    const float* pa = att + h * D;
    float acc = 0.f;
    for (int k = 0; k < D; ++k) {
        float v = pl[k] + pr[k];
        v = (v > 0.f) ? v : v * SLOPE;
        acc += pa[k] * v;
    }
    logit[idx] = acc;
    atomicMax(&mx[(size_t)d * H + h], fkey(acc));
}

__global__ void gat_edge_exp(const int* __restrict__ dst, int E, int nloops, int H,
                             const unsigned* __restrict__ mx,
                             float* __restrict__ logit, float* __restrict__ den)
{
    int idx = blockIdx.x * blockDim.x + threadIdx.x;
    int total = (E + nloops) * H;
    if (idx >= total) return;
    int e = idx / H, h = idx % H;
    int d = (e < E) ? dst[e] : (e - E);
    float ex = expf(logit[idx] - funkey(mx[(size_t)d * H + h]));
    logit[idx] = ex;
    atomicAdd(&den[(size_t)d * H + h], ex);
}

__global__ void gat_edge_msg(const int* __restrict__ src, const int* __restrict__ dst,
                             int E, int nloops, int H, int D,
                             const float* __restrict__ xl, const float* __restrict__ ex,
                             const float* __restrict__ den, float* __restrict__ agg)
{
    int idx = blockIdx.x * blockDim.x + threadIdx.x;
    int total = (E + nloops) * H;
    if (idx >= total) return;
    int e = idx / H, h = idx % H;
    int s = (e < E) ? src[e] : (e - E);
    int d = (e < E) ? dst[e] : (e - E);
    int HD = H * D;
    float alpha = ex[idx] / den[(size_t)d * H + h];
    const float* pl = xl + (size_t)s * HD + h * D;
    float* pa = agg + (size_t)d * HD + h * D;
    for (int k = 0; k < D; ++k)
        atomicAdd(&pa[k], pl[k] * alpha);
}

__global__ void gat_finalize(const float* __restrict__ agg, const float* __restrict__ cnt,
                             const float* __restrict__ bias, float* __restrict__ out,
                             int N, int C, int relu)
{
    int idx = blockIdx.x * blockDim.x + threadIdx.x;
    if (idx >= N * C) return;
    int n = idx / C, c = idx % C;
    float v = agg[idx] / fmaxf(cnt[n], 1.0f) + bias[c];
    if (relu) v = fmaxf(v, 0.f);
    out[idx] = v;
}

// ----------------------------- misc -----------------------------------------
__global__ void concat_zh(const float* __restrict__ z, const float* __restrict__ hh,
                          float* __restrict__ out, int N)
{
    int idx = blockIdx.x * blockDim.x + threadIdx.x;
    if (idx >= N * 33) return;
    int n = idx / 33, c = idx % 33;
    out[idx] = (c == 0) ? z[n] : hh[(size_t)n * 32 + (c - 1)];
}

__global__ void pred_final_k(const float* __restrict__ h, const float* __restrict__ w,
                             const float* __restrict__ b, float* __restrict__ out, int N)
{
    int n = blockIdx.x * blockDim.x + threadIdx.x;
    if (n >= N) return;
    float acc = b[0];
    for (int k = 0; k < 16; ++k) acc += h[(size_t)n * 16 + k] * w[k];
    out[n] = acc;
}

// ----------------------------- host-side helpers -----------------------------
struct GatP { const float *Wl, *Wr, *att, *bias, *bl, *br; };

static void gat_layer(hipStream_t s,
                      const float* Xsrc, int Ksrc, const float* Xdst, int Kdst,
                      int n_src, int n_dst,
                      const int* esrc, const int* edst, int E, int nloops,
                      const GatP& g, int H, int D,
                      const float* cnt, float* out, int relu,
                      float* xl, float* xr, float* agg, float* logit,
                      float* den, unsigned* mx)
{
    int HD = H * D;
    int bthr = 32 * (HD / 16);   // one wave per 16-col tile, shared A strip
    gemm16_wmma<<<CDIV(n_src, 16), bthr, 0, s>>>(Xsrc, g.Wl, g.bl, xl, n_src, Ksrc, HD, 0);
    gemm16_wmma<<<CDIV(n_dst, 16), bthr, 0, s>>>(Xdst, g.Wr, g.br, xr, n_dst, Kdst, HD, 0);
    hipMemsetAsync(agg, 0, (size_t)n_dst * HD * sizeof(float), s);
    hipMemsetAsync(den, 0, (size_t)n_dst * H * sizeof(float), s);
    hipMemsetAsync(mx, 0, (size_t)n_dst * H * sizeof(unsigned), s);
    int tot = (E + nloops) * H;
    gat_edge_logit<<<CDIV(tot, 256), 256, 0, s>>>(esrc, edst, E, nloops, xl, xr,
                                                  g.att, H, D, logit, mx);
    gat_edge_exp<<<CDIV(tot, 256), 256, 0, s>>>(edst, E, nloops, H, mx, logit, den);
    gat_edge_msg<<<CDIV(tot, 256), 256, 0, s>>>(esrc, edst, E, nloops, H, D,
                                                xl, logit, den, agg);
    gat_finalize<<<CDIV(n_dst * HD, 256), 256, 0, s>>>(agg, cnt, g.bias, out,
                                                       n_dst, HD, relu);
}

// ----------------------------- launcher --------------------------------------
// Input flattening assumption: top-level dict in insertion order
// (x_low, z_high, params, edge_up, edge_uh, edge_high); nested `params` pytree
// flattened JAX-style: dict keys sorted, lists in index order.
//   params: bn_h0{beta,gamma}, bn_high[4]{beta,gamma}, bn_up[4]{beta,gamma},
//           cnn1d[3]{b,beta,gamma,w}, cnn2d[3]{b,beta,gamma,w},
//           gat_down{Wl,Wr,att,bias,bl,br}, gat_high[5]{...}, gat_up[5]{...},
//           pred[3]{b,w}
extern "C" void kernel_launch(void* const* d_in, const int* in_sizes, int n_in,
                              void* d_out, int out_size, void* d_ws, size_t ws_size,
                              hipStream_t stream)
{
    (void)in_sizes; (void)n_in; (void)out_size; (void)ws_size;
    int i = 0;
    const float* x_low  = (const float*)d_in[i++];
    const float* z_high = (const float*)d_in[i++];

    const float* bnh0_beta  = (const float*)d_in[i++];
    const float* bnh0_gamma = (const float*)d_in[i++];
    const float *bnhi_beta[4], *bnhi_gamma[4];
    for (int l = 0; l < 4; ++l) { bnhi_beta[l] = (const float*)d_in[i++]; bnhi_gamma[l] = (const float*)d_in[i++]; }
    const float *bnup_beta[4], *bnup_gamma[4];
    for (int l = 0; l < 4; ++l) { bnup_beta[l] = (const float*)d_in[i++]; bnup_gamma[l] = (const float*)d_in[i++]; }

    struct CnnP { const float *b, *beta, *gamma, *w; };
    CnnP c1d[3], c2d[3];
    for (int l = 0; l < 3; ++l) {
        c1d[l].b = (const float*)d_in[i++]; c1d[l].beta = (const float*)d_in[i++];
        c1d[l].gamma = (const float*)d_in[i++]; c1d[l].w = (const float*)d_in[i++];
    }
    for (int l = 0; l < 3; ++l) {
        c2d[l].b = (const float*)d_in[i++]; c2d[l].beta = (const float*)d_in[i++];
        c2d[l].gamma = (const float*)d_in[i++]; c2d[l].w = (const float*)d_in[i++];
    }
    auto read_gat = [&](GatP& g) {
        g.Wl = (const float*)d_in[i++]; g.Wr = (const float*)d_in[i++];
        g.att = (const float*)d_in[i++]; g.bias = (const float*)d_in[i++];
        g.bl = (const float*)d_in[i++]; g.br = (const float*)d_in[i++];
    };
    GatP gdown; read_gat(gdown);
    GatP ghigh[5]; for (int l = 0; l < 5; ++l) read_gat(ghigh[l]);
    GatP gup[5];   for (int l = 0; l < 5; ++l) read_gat(gup[l]);
    const float *pred_b[3], *pred_w[3];
    for (int l = 0; l < 3; ++l) { pred_b[l] = (const float*)d_in[i++]; pred_w[l] = (const float*)d_in[i++]; }

    const int* e_up   = (const int*)d_in[i++];
    const int* e_uh   = (const int*)d_in[i++];
    const int* e_high = (const int*)d_in[i++];

    // workspace layout (floats)
    float* Wf = (float*)d_ws;
    const size_t NB = (size_t)N_UP * 64;
    float* A    = Wf;
    float* B    = A + NB;
    float* XLb  = B + NB;
    float* XRb  = XLb + NB;
    float* AGG  = XRb + NB;
    float* LOGb = AGG + NB;                       // 1.2M
    unsigned* MX = (unsigned*)(LOGb + 1200000);   // 0.2M
    float* DEN  = (float*)(MX + 200000);          // 0.2M
    float* CNTU = DEN + 200000;
    float* CNTH = CNTU + N_UP;
    float* CNTG = CNTH + N_HIGH;
    float* MEAN = CNTG + N_HIGH;
    float* RSTD = MEAN + 64;

    // ---- degrees (recomputed every call; no cross-call state) ----
    hipMemsetAsync(CNTU, 0, N_UP * sizeof(float), stream);
    hipMemsetAsync(CNTH, 0, N_HIGH * sizeof(float), stream);
    hipMemsetAsync(CNTG, 0, N_HIGH * sizeof(float), stream);
    degree_k<<<CDIV(E_UP, 256), 256, 0, stream>>>(e_up + E_UP, E_UP, 0, CNTU);
    degree_k<<<CDIV(E_UH, 256), 256, 0, stream>>>(e_uh + E_UH, E_UH, 0, CNTH);
    degree_k<<<CDIV(E_HIGH + N_HIGH, 256), 256, 0, stream>>>(e_high + E_HIGH, E_HIGH, N_HIGH, CNTG);

    // ---- CNN2D: 3x (depthwise conv + BN + ReLU) ----
    const float* cur = x_low;
    for (int l = 0; l < 3; ++l) {
        dwconv2d_k<<<CDIV(N_LOW * 125, 256), 256, 0, stream>>>(cur, c2d[l].w, c2d[l].b, A, N_LOW);
        bn_stats_strided<<<5, 256, 0, stream>>>(A, N_LOW, 5, 25, MEAN, RSTD);
        bn_apply_strided<<<CDIV(N_LOW * 125, 256), 256, 0, stream>>>(A, B, N_LOW, 5, 25,
                                                                     MEAN, RSTD, c2d[l].gamma, c2d[l].beta, 1);
        cur = B;
    }
    maxpool2d_k<<<CDIV(N_LOW * 45, 256), 256, 0, stream>>>(B, A, N_LOW);

    // ---- CNN1D: 3x (conv1d + BN + ReLU), maxpool1d ----
    int ci_arr[3] = {1, 5, 25}, co_arr[3] = {5, 25, 25};
    const float* c1in = A;
    for (int l = 0; l < 3; ++l) {
        int Ci = ci_arr[l], Co = co_arr[l];
        conv1d_k<<<CDIV(N_LOW * Co * 45, 256), 256, 0, stream>>>(c1in, c1d[l].w, c1d[l].b, B,
                                                                 N_LOW, Ci, Co, 45);
        bn_stats_strided<<<Co, 256, 0, stream>>>(B, N_LOW, Co, 45, MEAN, RSTD);
        bn_apply_strided<<<CDIV(N_LOW * Co * 45, 256), 256, 0, stream>>>(B, A, N_LOW, Co, 45,
                                                                         MEAN, RSTD, c1d[l].gamma, c1d[l].beta, 1);
        c1in = A;
    }
    maxpool1d_k<<<CDIV(N_UP * 23, 256), 256, 0, stream>>>(A, B, N_LOW);   // B = [N_UP,23]

    // ---- processor_upscaled: 4x (GATv2 H=2 + BN + ReLU) + GATv2 H=1 + ReLU ----
    const float* xin = B; int Kin = 23;
    for (int l = 0; l < 4; ++l) {
        gat_layer(stream, xin, Kin, xin, Kin, N_UP, N_UP,
                  e_up, e_up + E_UP, E_UP, 0, gup[l], 2, HIDV,
                  CNTU, A, 0, XLb, XRb, AGG, LOGb, DEN, MX);
        bn_stats_strided<<<64, 256, 0, stream>>>(A, N_UP, 64, 1, MEAN, RSTD);
        bn_apply_strided<<<CDIV(N_UP * 64, 256), 256, 0, stream>>>(A, B, N_UP, 64, 1,
                                                                   MEAN, RSTD, bnup_gamma[l], bnup_beta[l], 1);
        xin = B; Kin = 64;
    }
    gat_layer(stream, B, 64, B, 64, N_UP, N_UP,
              e_up, e_up + E_UP, E_UP, 0, gup[4], 1, HIDV,
              CNTU, A, 1, XLb, XRb, AGG, LOGb, DEN, MX);     // A = [N_UP,32]

    // ---- downscaler (bipartite, Kdst=1 handled by padded WMMA GEMM) ----
    gat_layer(stream, A, 32, z_high, 1, N_UP, N_HIGH,
              e_uh, e_uh + E_UH, E_UH, 0, gdown, 1, HIDV,
              CNTH, B, 0, XLb, XRb, AGG, LOGb, DEN, MX);     // B = [N_HIGH,32]
    concat_zh<<<CDIV(N_HIGH * 33, 256), 256, 0, stream>>>(z_high, B, A, N_HIGH);
    bn_stats_strided<<<33, 256, 0, stream>>>(A, N_HIGH, 33, 1, MEAN, RSTD);
    bn_apply_strided<<<CDIV(N_HIGH * 33, 256), 256, 0, stream>>>(A, B, N_HIGH, 33, 1,
                                                                 MEAN, RSTD, bnh0_gamma, bnh0_beta, 0);

    // ---- processor_high (with self loops) ----
    xin = B; Kin = 33;
    for (int l = 0; l < 4; ++l) {
        gat_layer(stream, xin, Kin, xin, Kin, N_HIGH, N_HIGH,
                  e_high, e_high + E_HIGH, E_HIGH, N_HIGH, ghigh[l], 2, HIDV,
                  CNTG, A, 0, XLb, XRb, AGG, LOGb, DEN, MX);
        bn_stats_strided<<<64, 256, 0, stream>>>(A, N_HIGH, 64, 1, MEAN, RSTD);
        bn_apply_strided<<<CDIV(N_HIGH * 64, 256), 256, 0, stream>>>(A, B, N_HIGH, 64, 1,
                                                                     MEAN, RSTD, bnhi_gamma[l], bnhi_beta[l], 1);
        xin = B; Kin = 64;
    }
    gat_layer(stream, B, 64, B, 64, N_HIGH, N_HIGH,
              e_high, e_high + E_HIGH, E_HIGH, N_HIGH, ghigh[4], 1, HIDV,
              CNTG, A, 1, XLb, XRb, AGG, LOGb, DEN, MX);     // A = [N_HIGH,32]

    // ---- predictor MLP: 32->32 (relu) -> 16 (relu) -> 1 ----
    gemm16_wmma<<<CDIV(N_HIGH, 16), 64, 0, stream>>>(A, pred_w[0], pred_b[0], B, N_HIGH, 32, 32, 1);
    gemm16_wmma<<<CDIV(N_HIGH, 16), 32, 0, stream>>>(B, pred_w[1], pred_b[1], XLb, N_HIGH, 32, 16, 1);
    pred_final_k<<<CDIV(N_HIGH, 256), 256, 0, stream>>>(XLb, pred_w[2], pred_b[2],
                                                        (float*)d_out, N_HIGH);
}
